// RetNetAdapter_85581518340432
// MI455X (gfx1250) — matrix operand order repaired and checked by reading, not compiled, not run on hardware
//
#include <hip/hip_runtime.h>
#include <hip/hip_bf16.h>

typedef __bf16 bf16;
typedef __attribute__((ext_vector_type(8)))  __bf16 v8bf;
typedef __attribute__((ext_vector_type(16))) __bf16 v16bf;
typedef __attribute__((ext_vector_type(8)))  float  v8f;

#define D_MODEL 768
#define D_INNER 192
#define SEQ     32
#define HW      196
#define BATCH   8
#define NSEQ    (BATCH * HW)   // 1568 sequences

// ---------------- fragment loaders (bf16, 16x16x32 WMMA) ----------------
// A: 16x32 tile, row-major source with leading dim `ld` (elements).
// Lane L holds row M=L&15; elements 0..7 -> K = half*8 + e, 8..15 -> K = 16 + half*8 + (e-8).
__device__ __forceinline__ v16bf load_frag_A(const bf16* tile, int ld, int lane) {
    const int r = lane & 15, half = (lane >> 4) & 1;
    const bf16* p = tile + r * ld + half * 8;
    v8bf lo = *reinterpret_cast<const v8bf*>(p);
    v8bf hi = *reinterpret_cast<const v8bf*>(p + 16);
    return __builtin_shufflevector(lo, hi, 0,1,2,3,4,5,6,7,8,9,10,11,12,13,14,15);
}

// B: 32x16 tile (KxN). Source stored N-major (row = output column, K contiguous),
// leading dim `ld`. Lane L holds column N=L&15; elements 0..7 -> K = half*8 + e, 8..15 -> K = 16 + half*8 + (e-8).
__device__ __forceinline__ v16bf load_frag_B(const bf16* tileT, int ld, int lane) {
    const int n = lane & 15, half = (lane >> 4) & 1;
    const bf16* p = tileT + n * ld + half * 8;
    v8bf lo = *reinterpret_cast<const v8bf*>(p);
    v8bf hi = *reinterpret_cast<const v8bf*>(p + 16);
    return __builtin_shufflevector(lo, hi, 0,1,2,3,4,5,6,7,8,9,10,11,12,13,14,15);
}

__device__ __forceinline__ v8f wmma_bf16(v16bf a, v16bf b, v8f c) {
    return __builtin_amdgcn_wmma_f32_16x16x32_bf16(false, a, false, b, (short)0, c, false, false);
}

// ---------------- prep kernels ----------------
// W: KxN row-major f32  ->  Wt: NxK row-major bf16 (so B-fragments are K-contiguous)
__global__ void prep_transpose(const float* __restrict__ W, bf16* __restrict__ Wt, int K, int N) {
    int idx = blockIdx.x * blockDim.x + threadIdx.x;
    if (idx < K * N) {
        int k = idx / N, n = idx % N;
        Wt[n * K + k] = (bf16)W[idx];
    }
}

__global__ void prep_tables(float* __restrict__ cosQ, float* __restrict__ sinQ,
                            float* __restrict__ cosK, float* __restrict__ sinK,
                            float* __restrict__ decay) {
    int idx = blockIdx.x * blockDim.x + threadIdx.x;
    if (idx < SEQ * D_INNER) {
        int s = idx / D_INNER, dim = idx % D_INNER;
        int i = dim >> 1;
        const float Dh = (float)D_INNER;
        float base  = (2.0f * (float)i + 0.4f * Dh) / (1.4f * Dh);
        float power = (float)s / 512.0f;
        float scale = powf(base, power);
        float invf  = powf(10000.0f, -((float)i) / (Dh * 0.5f));
        float ang   = (float)s * invf;
        float sn = sinf(ang), cs = cosf(ang);
        cosQ[idx] = cs * scale;
        sinQ[idx] = sn * scale;
        float is = 1.0f / scale;
        cosK[idx] = cs * is;
        sinK[idx] = sn * is;
    }
    if (idx < SEQ * SEQ) {
        int nq = idx / SEQ, mk = idx % SEQ;
        decay[idx] = (nq >= mk) ? powf(0.5f, (float)(nq - mk)) : 0.0f;
    }
}

// ---------------- fused main kernel: one block per sequence ----------------
__global__ __launch_bounds__(256) void retnet_fused(
    const float* __restrict__ x,
    const float* __restrict__ b1, const float* __restrict__ b2,
    const float* __restrict__ ln_g, const float* __restrict__ ln_b,
    const bf16* __restrict__ W1t, const bf16* __restrict__ WQt,
    const bf16* __restrict__ WKt, const bf16* __restrict__ WVt,
    const bf16* __restrict__ W2t,
    const float* __restrict__ cosQ, const float* __restrict__ sinQ,
    const float* __restrict__ cosK, const float* __restrict__ sinK,
    const float* __restrict__ decay,
    float* __restrict__ out)
{
    __shared__ __align__(16) bf16  x_bf [SEQ * D_MODEL];   // 48 KB
    __shared__ __align__(16) float hf   [SEQ * D_INNER];   // 24 KB (f32 x_down for exact LN)
    __shared__ __align__(16) bf16  h_bf [SEQ * D_INNER];   // 12 KB
    __shared__ __align__(16) bf16  Qs   [SEQ * D_INNER];   // 12 KB
    __shared__ __align__(16) bf16  Ks   [SEQ * D_INNER];   // 12 KB
    __shared__ __align__(16) bf16  Vt   [D_INNER * SEQ];   // 12 KB (transposed V)
    __shared__ __align__(16) bf16  att_bf[SEQ * SEQ];      //  2 KB
    __shared__ __align__(16) bf16  ret_bf[SEQ * D_INNER];  // 12 KB

    const int tid  = threadIdx.x;
    const int lane = tid & 31;
    const int w    = tid >> 5;          // wave id 0..7

    const int seqId = blockIdx.x;       // bi*HW + hi
    const int bi = seqId / HW, hi = seqId % HW;
    const size_t rowbase = ((size_t)hi * (BATCH * SEQ) + (size_t)bi * SEQ) * D_MODEL;
    const float* xs = x   + rowbase;    // contiguous 32x768 f32 tile
    float*       os = out + rowbase;

    // warm L2 for the last (biggest) weight while we work
    for (int off = tid; off < (D_MODEL * D_INNER) / 64; off += 256)
        __builtin_prefetch(W2t + (size_t)off * 64, 0, 1);

    // ---- stage x tile -> bf16 LDS ----
    for (int idx = tid; idx < SEQ * D_MODEL / 4; idx += 256) {
        const float4 v = reinterpret_cast<const float4*>(xs)[idx];
        bf16* d = x_bf + idx * 4;
        d[0] = (bf16)v.x; d[1] = (bf16)v.y; d[2] = (bf16)v.z; d[3] = (bf16)v.w;
    }
    __syncthreads();

    // ---- GEMM1: x_down = X(32x768) @ W1(768x192) + b1  -> hf (f32) ----
    for (int t = w; t < 24; t += 8) {                 // 2 Mtiles x 12 Ntiles
        const int mt = t / 12, nt = t % 12;
        v8f acc = {};
        #pragma unroll
        for (int kk = 0; kk < 24; ++kk) {
            v16bf a  = load_frag_A(x_bf + mt * 16 * D_MODEL + kk * 32, D_MODEL, lane);
            v16bf bb = load_frag_B(W1t  + nt * 16 * D_MODEL + kk * 32, D_MODEL, lane);
            acc = wmma_bf16(a, bb, acc);
        }
        const int mb = mt * 16 + ((lane >> 4) & 1) * 8;
        const int n  = nt * 16 + (lane & 15);
        const float bias = b1[n];
        #pragma unroll
        for (int r = 0; r < 8; ++r)
            hf[(mb + r) * D_INNER + n] = acc[r] + bias;
    }
    __syncthreads();

    // ---- LayerNorm over d=192 per row ----
    if (tid < SEQ) {
        const float* row = hf + tid * D_INNER;
        float mu = 0.0f;
        for (int c = 0; c < D_INNER; ++c) mu += row[c];
        mu *= (1.0f / D_INNER);
        float var = 0.0f;
        for (int c = 0; c < D_INNER; ++c) { float dd = row[c] - mu; var += dd * dd; }
        var *= (1.0f / D_INNER);
        const float rstd = rsqrtf(var + 1e-5f);
        for (int c = 0; c < D_INNER; ++c)
            h_bf[tid * D_INNER + c] = (bf16)(((row[c] - mu) * rstd) * ln_g[c] + ln_b[c]);
    }
    __syncthreads();

    // ---- Q/K/V = h(32x192) @ W(192x192); xPos on Q,K in epilogue ----
    for (int t = w; t < 72; t += 8) {                 // 3 mats x 24 tiles
        const int which = t / 24, tt = t % 24;
        const int mt = tt / 12, nt = tt % 12;
        const bf16* Wsel = (which == 0) ? WQt : ((which == 1) ? WKt : WVt);
        v8f acc = {};
        #pragma unroll
        for (int kk = 0; kk < 6; ++kk) {
            v16bf a  = load_frag_A(h_bf + mt * 16 * D_INNER + kk * 32, D_INNER, lane);
            v16bf bb = load_frag_B(Wsel + nt * 16 * D_INNER + kk * 32, D_INNER, lane);
            acc = wmma_bf16(a, bb, acc);
        }
        const int mb = mt * 16 + ((lane >> 4) & 1) * 8;
        const int n  = nt * 16 + (lane & 15);
        if (which == 2) {
            #pragma unroll
            for (int r = 0; r < 8; ++r)
                Vt[n * SEQ + (mb + r)] = (bf16)acc[r];   // store V transposed
        } else {
            const float* cs = (which == 0) ? cosQ : cosK;
            const float* sn = (which == 0) ? sinQ : sinK;
            bf16* dst = (which == 0) ? Qs : Ks;
            #pragma unroll
            for (int r = 0; r < 8; ++r) {
                const int m = mb + r;
                const float v = acc[r];
                const float partner = __shfl_xor(v, 1, 32);  // (m, n^1) lives in lane^1
                const float rot = (n & 1) ? partner : -partner;
                dst[m * D_INNER + n] =
                    (bf16)(v * cs[m * D_INNER + n] + rot * sn[m * D_INNER + n]);
            }
        }
    }
    __syncthreads();

    // ---- att = (Q Kt) * decay  (32x32, K=192) ----
    if (w < 4) {
        const int mt = (w >> 1) & 1, nt = w & 1;
        v8f acc = {};
        #pragma unroll
        for (int kk = 0; kk < 6; ++kk) {
            v16bf a  = load_frag_A(Qs + mt * 16 * D_INNER + kk * 32, D_INNER, lane);
            // rows of K are output columns -> B loader applies directly
            v16bf bb = load_frag_B(Ks + nt * 16 * D_INNER + kk * 32, D_INNER, lane);
            acc = wmma_bf16(a, bb, acc);
        }
        const int mb = mt * 16 + ((lane >> 4) & 1) * 8;
        const int n  = nt * 16 + (lane & 15);
        #pragma unroll
        for (int r = 0; r < 8; ++r) {
            const int m = mb + r;
            att_bf[m * SEQ + n] = (bf16)(acc[r] * decay[m * SEQ + n]);
        }
    }
    __syncthreads();

    // ---- ret = att(32x32) @ V(32x192)  (single K step) ----
    for (int t = w; t < 24; t += 8) {
        const int mt = t / 12, nt = t % 12;
        v16bf a  = load_frag_A(att_bf + mt * 16 * SEQ, SEQ, lane);
        v16bf bb = load_frag_B(Vt     + nt * 16 * SEQ, SEQ, lane);
        v8f acc = {};
        acc = wmma_bf16(a, bb, acc);
        const int mb = mt * 16 + ((lane >> 4) & 1) * 8;
        const int n  = nt * 16 + (lane & 15);
        #pragma unroll
        for (int r = 0; r < 8; ++r)
            ret_bf[(mb + r) * D_INNER + n] = (bf16)acc[r];
    }
    __syncthreads();

    // ---- out = X + ret(32x192) @ W2(192x768) + b2 ----
    for (int t = w; t < 96; t += 8) {                 // 2 Mtiles x 48 Ntiles
        const int mt = t / 48, nt = t % 48;
        v8f acc = {};
        #pragma unroll
        for (int kk = 0; kk < 6; ++kk) {
            v16bf a  = load_frag_A(ret_bf + mt * 16 * D_INNER + kk * 32, D_INNER, lane);
            v16bf bb = load_frag_B(W2t    + nt * 16 * D_INNER + kk * 32, D_INNER, lane);
            acc = wmma_bf16(a, bb, acc);
        }
        const int mb = mt * 16 + ((lane >> 4) & 1) * 8;
        const int n  = nt * 16 + (lane & 15);
        const float bias = b2[n];
        #pragma unroll
        for (int r = 0; r < 8; ++r) {
            const int m = mb + r;
            os[(size_t)m * D_MODEL + n] = acc[r] + bias + xs[(size_t)m * D_MODEL + n];
        }
    }
}

// ---------------- launch ----------------
extern "C" void kernel_launch(void* const* d_in, const int* in_sizes, int n_in,
                              void* d_out, int out_size, void* d_ws, size_t ws_size,
                              hipStream_t stream) {
    const float* x    = (const float*)d_in[0];
    const float* W1   = (const float*)d_in[1];
    const float* b1   = (const float*)d_in[2];
    const float* W2   = (const float*)d_in[3];
    const float* b2   = (const float*)d_in[4];
    const float* ln_g = (const float*)d_in[5];
    const float* ln_b = (const float*)d_in[6];
    const float* WQ   = (const float*)d_in[7];
    const float* WK   = (const float*)d_in[8];
    const float* WV   = (const float*)d_in[9];

    char* ws = (char*)d_ws;
    bf16*  W1t  = (bf16*)(ws + 0);        // 768*192 bf16 = 294912 B
    bf16*  W2t  = (bf16*)(ws + 294912);   // 294912 B
    bf16*  WQt  = (bf16*)(ws + 589824);   // 73728 B
    bf16*  WKt  = (bf16*)(ws + 663552);
    bf16*  WVt  = (bf16*)(ws + 737280);
    float* cosQ = (float*)(ws + 811008);  // 32*192 f32 = 24576 B each
    float* sinQ = (float*)(ws + 835584);
    float* cosK = (float*)(ws + 860160);
    float* sinK = (float*)(ws + 884736);
    float* dec  = (float*)(ws + 909312);  // 32*32 f32 = 4096 B

    const int TP = 256;
    prep_transpose<<<(768 * 192 + TP - 1) / TP, TP, 0, stream>>>(W1, W1t, 768, 192);
    prep_transpose<<<(192 * 768 + TP - 1) / TP, TP, 0, stream>>>(W2, W2t, 192, 768);
    prep_transpose<<<(192 * 192 + TP - 1) / TP, TP, 0, stream>>>(WQ, WQt, 192, 192);
    prep_transpose<<<(192 * 192 + TP - 1) / TP, TP, 0, stream>>>(WK, WKt, 192, 192);
    prep_transpose<<<(192 * 192 + TP - 1) / TP, TP, 0, stream>>>(WV, WVt, 192, 192);
    prep_tables<<<(SEQ * D_INNER + TP - 1) / TP, TP, 0, stream>>>(cosQ, sinQ, cosK, sinK, dec);

    retnet_fused<<<NSEQ, 256, 0, stream>>>(x, b1, b2, ln_g, ln_b,
                                           W1t, WQt, WKt, WVt, W2t,
                                           cosQ, sinQ, cosK, sinK, dec,
                                           (float*)d_out);
}
